// MoLoRALinear_80728205295877
// MI455X (gfx1250) — matrix operand order repaired and checked by reading, not compiled, not run on hardware
//
#include <hip/hip_runtime.h>

// Problem dims (fixed by reference): B=2,S=2048 -> M=4096 tokens; H=2048; O=2048; E=8; R=16.
// KP = H + E*R = 2176 (base K extended with LoRA rank columns).
#define MTOK 4096
#define HDIM 2048
#define ODIM 2048
#define NEXP 8
#define RANK 16
#define KP   2176

typedef __attribute__((ext_vector_type(16))) __bf16 bf16x16;
typedef __attribute__((ext_vector_type(8)))  float  f32x8;

__device__ __forceinline__ unsigned short f2bf(float f) {
  union { float f; unsigned u; } v; v.f = f;
  unsigned u = v.u;
  u += 0x7fffu + ((u >> 16) & 1u);   // round-to-nearest-even
  return (unsigned short)(u >> 16);
}

// CDNA5 async DMA: global -> LDS, 16B per lane, tracked with ASYNCcnt.
// Flat-address low 32 bits are the LDS byte offset (ISA 10.2 aperture mapping).
__device__ __forceinline__ void async_g2l_b128(unsigned long long g, unsigned l) {
  asm volatile("global_load_async_to_lds_b128 %0, %1, off"
               :: "v"(l), "v"(g) : "memory");
}
__device__ __forceinline__ void wait_async0() {
  asm volatile("s_wait_asynccnt 0x0" ::: "memory");
}

// ---------------------------------------------------------------------------
// Pack kernels: fp32 -> bf16 operand buffers
// ---------------------------------------------------------------------------

// Xp[m, 0:2048) = bf16(x[m, :]); Xp[m, 2048:2176) = 0 (filled later by router)
__global__ __launch_bounds__(256) void pack_x_kernel(
    const float* __restrict__ x, unsigned short* __restrict__ Xp) {
  long idx = (long)blockIdx.x * 256 + threadIdx.x;    // [0, 4096*2176)
  long m = idx / KP, k = idx % KP;
  float f = (k < HDIM) ? x[m * HDIM + k] : 0.0f;
  Xp[idx] = f2bf(f);
}

// Wp[o, 0:2048) = bf16(W[o, :]); Wp[o, 2048+e*16+r) = bf16(alpha * B_w[e,o,r])
__global__ __launch_bounds__(256) void pack_weight_kernel(
    const float* __restrict__ W, const float* __restrict__ Bw,
    unsigned short* __restrict__ Wp) {
  long idx = (long)blockIdx.x * 256 + threadIdx.x;    // [0, 2048*2176)
  long o = idx / KP, k = idx % KP;
  float f;
  if (k < HDIM) {
    f = W[o * HDIM + k];
  } else {
    long c = k - HDIM;
    long e = c >> 4, r = c & 15;
    f = 16.0f * Bw[(e * (long)ODIM + o) * RANK + r];  // LORA_ALPHA folded here
  }
  Wp[idx] = f2bf(f);
}

// A_w flat is exactly [E*R=128, H] row-major -> straight convert
__global__ __launch_bounds__(256) void pack_acat_kernel(
    const float* __restrict__ Aw, unsigned short* __restrict__ Acat) {
  long idx = (long)blockIdx.x * 256 + threadIdx.x;    // [0, 128*2048)
  Acat[idx] = f2bf(Aw[idx]);
}

// ---------------------------------------------------------------------------
// Tiled bf16 WMMA GEMM:  C[M,N] = A[M,K] * B[N,K]^T   (both K-contiguous)
// block = 256 threads (8 wave32), block tile 128x128, K-step 32.
// Double-buffered LDS filled by global_load_async_to_lds_b128 (ASYNCcnt),
// loop-carried 64-bit global cursors (one v_add_nc_u64 each per step).
// Per wave: 64x32 output = 4x2 tiles of v_wmma_f32_16x16x32_bf16.
// ---------------------------------------------------------------------------
#define BM 128
#define BN 128
#define BK 32
#define LDSP 40                    // LDS row pitch in u16 (80B, 16B aligned)
#define BUFB (BM * LDSP * 2)       // bytes per LDS buffer (10240)

__global__ __launch_bounds__(256) void wmma_gemm_bf16(
    const unsigned short* __restrict__ A, int lda,
    const unsigned short* __restrict__ B, int ldb,
    float* __restrict__ C, int ldc, int K) {
  __shared__ unsigned short sA[2][BM * LDSP];
  __shared__ unsigned short sB[2][BN * LDSP];

  const int tid  = threadIdx.x;
  const int lane = tid & 31;
  const int wave = tid >> 5;
  const int wm   = wave & 1;     // 0..1 -> 64-row slab
  const int wn   = wave >> 1;    // 0..3 -> 32-col slab
  const long bm  = (long)blockIdx.y * BM;
  const long bn  = (long)blockIdx.x * BN;

  // staging map: each thread DMAs two 16B chunks per tile per operand
  const int crow = tid >> 2;        // 0..63
  const int ccol = (tid & 3) * 8;   // 0,8,16,24 u16

  // loop-carried global cursors (bytes), advanced by BK*2 per K-step
  unsigned long long gA0 = (unsigned long long)&A[(bm + crow) * (long)lda + ccol];
  unsigned long long gA1 = gA0 + 64ull * (unsigned long long)lda * 2ull;
  unsigned long long gB0 = (unsigned long long)&B[(bn + crow) * (long)ldb + ccol];
  unsigned long long gB1 = gB0 + 64ull * (unsigned long long)ldb * 2ull;

  // fixed LDS destinations (buffer select = +BUFB)
  const unsigned lA = (unsigned)(unsigned long long)&sA[0][crow * LDSP + ccol];
  const unsigned lB = (unsigned)(unsigned long long)&sB[0][crow * LDSP + ccol];
  const unsigned lOfs64 = 64u * LDSP * 2u;   // second 64-row slab of a tile

  const f32x8 zero = {0.f, 0.f, 0.f, 0.f, 0.f, 0.f, 0.f, 0.f};
  f32x8 acc[4][2];
#pragma unroll
  for (int i = 0; i < 4; ++i)
#pragma unroll
    for (int j = 0; j < 2; ++j) acc[i][j] = zero;

  const int arow = wm * 64 + (lane & 15);   // A-frag row within block tile
  const int brow = wn * 32 + (lane & 15);   // B-frag row (= output column)
  const int kh   = lane >> 4;               // K-half select per ISA layout

  const int nsteps = K / BK;

  // prologue: async-stage tile 0 into buffer 0, advance cursors
  async_g2l_b128(gA0, lA);
  async_g2l_b128(gA1, lA + lOfs64);
  async_g2l_b128(gB0, lB);
  async_g2l_b128(gB1, lB + lOfs64);
  gA0 += BK * 2; gA1 += BK * 2; gB0 += BK * 2; gB1 += BK * 2;
  wait_async0();
  __syncthreads();

  for (int s = 0; s < nsteps; ++s) {
    const int buf = s & 1;

    // async-stage next tile into the other buffer (no wait yet)
    if (s + 1 < nsteps) {
      const unsigned bo = (unsigned)(buf ^ 1) * BUFB;
      async_g2l_b128(gA0, lA + bo);
      async_g2l_b128(gA1, lA + bo + lOfs64);
      async_g2l_b128(gB0, lB + bo);
      async_g2l_b128(gB1, lB + bo + lOfs64);
      gA0 += BK * 2; gA1 += BK * 2; gB0 += BK * 2; gB1 += BK * 2;
      if (s + 2 < nsteps) {  // cursors now point 2 tiles ahead: warm near caches
        __builtin_prefetch((const void*)gA0, 0, 3);
        __builtin_prefetch((const void*)gB0, 0, 3);
      }
    }

    // A fragment (16x32 bf16): VGPR0-3 = K[kh*8 .. +7], VGPR4-7 = K[16+kh*8 .. +7]
    bf16x16 af[4];
#pragma unroll
    for (int i = 0; i < 4; ++i) {
      const unsigned short* p = &sA[buf][(arow + i * 16) * LDSP];
      union { uint4 u[2]; bf16x16 v; } f;
      f.u[0] = *(const uint4*)&p[kh * 8];
      f.u[1] = *(const uint4*)&p[16 + kh * 8];
      af[i] = f.v;
    }
    // B fragment (32x16 bf16): VGPR0-3 = K[kh*8 .. +7], VGPR4-7 = K[16+kh*8 .. +7]
    bf16x16 bfr[2];
#pragma unroll
    for (int j = 0; j < 2; ++j) {
      const unsigned short* p = &sB[buf][(brow + j * 16) * LDSP];
      union { uint4 u[2]; bf16x16 v; } f;
      f.u[0] = *(const uint4*)&p[kh * 8];
      f.u[1] = *(const uint4*)&p[16 + kh * 8];
      bfr[j] = f.v;
    }

#pragma unroll
    for (int i = 0; i < 4; ++i)
#pragma unroll
      for (int j = 0; j < 2; ++j)
        acc[i][j] = __builtin_amdgcn_wmma_f32_16x16x32_bf16(
            false, af[i], false, bfr[j], (short)0, acc[i][j], false, false);

    wait_async0();      // my DMA into buf^1 landed
    __syncthreads();    // whole block done writing buf^1 / reading buf
  }

  // C layout: lanes 0-15 -> M rows v, lanes 16-31 -> M rows v+8; N = lane&15
  const int mo = (lane >> 4) * 8;
  const int no = lane & 15;
#pragma unroll
  for (int i = 0; i < 4; ++i)
#pragma unroll
    for (int j = 0; j < 2; ++j) {
      long row = bm + wm * 64 + i * 16 + mo;
      long col = bn + wn * 32 + j * 16 + no;
#pragma unroll
      for (int v = 0; v < 8; ++v)
        C[(row + v) * (long)ldc + col] = acc[i][j][v];
    }
}

// ---------------------------------------------------------------------------
// Router: 1 wave32 per token. logits -> softmax -> top2 -> renorm -> scale T,
// write bf16 into the K-extension columns of Xp.
// ---------------------------------------------------------------------------
__global__ __launch_bounds__(256) void router_scale_kernel(
    const float* __restrict__ x,       // [4096, 2048] fp32
    const float* __restrict__ gate_w,  // [8, 2048] fp32
    const float* __restrict__ T,       // [4096, 128] fp32 (x @ Acat^T)
    unsigned short* __restrict__ Xp) { // [4096, 2176] bf16
  const int lane = threadIdx.x & 31;
  const int wave = threadIdx.x >> 5;
  const long m = (long)blockIdx.x * 8 + wave;

  float a[NEXP];
#pragma unroll
  for (int e = 0; e < NEXP; ++e) a[e] = 0.0f;
  for (int h = lane; h < HDIM; h += 32) {
    float xv = x[m * HDIM + h];
#pragma unroll
    for (int e = 0; e < NEXP; ++e) a[e] += xv * gate_w[e * HDIM + h];
  }
#pragma unroll
  for (int e = 0; e < NEXP; ++e)
    for (int off = 16; off >= 1; off >>= 1)
      a[e] += __shfl_xor(a[e], off, 32);

  // softmax over 8 experts (all lanes redundantly)
  float mx = a[0];
#pragma unroll
  for (int e = 1; e < NEXP; ++e) mx = fmaxf(mx, a[e]);
  float p[NEXP], s = 0.0f;
#pragma unroll
  for (int e = 0; e < NEXP; ++e) { p[e] = __expf(a[e] - mx); s += p[e]; }
#pragma unroll
  for (int e = 0; e < NEXP; ++e) p[e] /= s;

  int i1 = 0;
#pragma unroll
  for (int e = 1; e < NEXP; ++e) if (p[e] > p[i1]) i1 = e;
  int i2 = (i1 == 0) ? 1 : 0;
#pragma unroll
  for (int e = 0; e < NEXP; ++e) if (e != i1 && p[e] > p[i2]) i2 = e;
  float sw = p[i1] + p[i2];
  float w1 = p[i1] / sw, w2 = p[i2] / sw;

  // scale the 128 rank columns by cw[e] and pack into Xp tail
#pragma unroll
  for (int q = 0; q < 4; ++q) {
    int c = lane + q * 32;             // 0..127
    int e = c >> 4;
    float w = (e == i1) ? w1 : ((e == i2) ? w2 : 0.0f);
    float val = T[m * 128 + c] * w;
    Xp[m * (long)KP + HDIM + c] = f2bf(val);
  }
}

// ---------------------------------------------------------------------------
// Launch
// ---------------------------------------------------------------------------
extern "C" void kernel_launch(void* const* d_in, const int* in_sizes, int n_in,
                              void* d_out, int out_size, void* d_ws, size_t ws_size,
                              hipStream_t stream) {
  (void)in_sizes; (void)n_in; (void)out_size; (void)ws_size;
  const float* x      = (const float*)d_in[0];  // [2,2048,2048]
  const float* weight = (const float*)d_in[1];  // [2048,2048]
  const float* gate_w = (const float*)d_in[2];  // [8,2048]
  const float* A_w    = (const float*)d_in[3];  // [8,16,2048]
  const float* B_w    = (const float*)d_in[4];  // [8,2048,16]
  float* out = (float*)d_out;                   // [2,2048,2048] fp32

  // workspace layout (16B-aligned offsets)
  char* ws = (char*)d_ws;
  unsigned short* Xp   = (unsigned short*)(ws);                       // 4096*2176 bf16 = 17,825,792 B
  unsigned short* Wp   = (unsigned short*)(ws + 17825792);            // 2048*2176 bf16 =  8,912,896 B
  unsigned short* Acat = (unsigned short*)(ws + 17825792 + 8912896);  // 128*2048 bf16  =    524,288 B
  float*          T    = (float*)(ws + 17825792 + 8912896 + 524288);  // 4096*128 f32   =  2,097,152 B

  pack_x_kernel     <<<34816, 256, 0, stream>>>(x, Xp);
  pack_weight_kernel<<<17408, 256, 0, stream>>>(weight, B_w, Wp);
  pack_acat_kernel  <<< 1024, 256, 0, stream>>>(A_w, Acat);

  // LoRA down-projection: T = Xp[:, :2048] @ Acat^T   (M=4096, N=128, K=2048)
  wmma_gemm_bf16<<<dim3(1, MTOK / BM), 256, 0, stream>>>(Xp, KP, Acat, HDIM, T, 128, HDIM);

  // router + scale + pack into K-extension columns
  router_scale_kernel<<<MTOK / 8, 256, 0, stream>>>(x, gate_w, T, Xp);

  // fused GEMM: out = Xp[4096x2176] @ Wp[2048x2176]^T  (base + alpha*lora in one pass)
  wmma_gemm_bf16<<<dim3(ODIM / BN, MTOK / BM), 256, 0, stream>>>(Xp, KP, Wp, KP, out, ODIM, KP);
}
